// CrossFormerBlock_67242007987033
// MI455X (gfx1250) — matrix operand order repaired and checked by reading, not compiled, not run on hardware
//
#include <hip/hip_runtime.h>
#include <hip/hip_bf16.h>
#include <cmath>

// ---------------------------------------------------------------------------
// CrossFormer block on gfx1250 (MI455X): all four big GEMMs + window attention
// run through v_wmma_f32_16x16x32_f16 (wave32 WMMA), f32 accumulate.
// GEMM: 128x128 block tile, 8 waves (32x64 each), double-buffered LDS filled
// with CDNA5 async global->LDS copies (ASYNCcnt + s_wait_asynccnt), 1 barrier
// per K-step.  Cross-lane reductions use ds_swizzle immediate XOR patterns.
// ---------------------------------------------------------------------------

typedef _Float16 v16h __attribute__((ext_vector_type(16)));
typedef _Float16 v8h  __attribute__((ext_vector_type(8)));
typedef float    v8f  __attribute__((ext_vector_type(8)));

#define TOKENS   50176          // B * H * W = 64*28*28
#define CDIM     384
#define QS       19267584       // TOKENS * CDIM  (one q/k/v section, f16 elems)
#define NPAIRS   12288          // 1024 windows * 12 heads
#define SCALE_Q  0.17677669529663687f   // 32^-0.5

// ds_swizzle group-of-32: offset = {0, xor[14:10], or[9:5]=0, and[4:0]=0x1f}
#define SWZ_XOR(x, m) \
  __int_as_float(__builtin_amdgcn_ds_swizzle(__float_as_int(x), ((m) << 10) | 0x1f))

__device__ __forceinline__ v8f wmma32(v16h a, v16h b, v8f c) {
  return __builtin_amdgcn_wmma_f32_16x16x32_f16(false, a, false, b,
                                                (short)0, c, false, false);
}

// Async global -> LDS copy of 16 bytes per lane (GLOBAL_LOAD_ASYNC_TO_LDS_B128,
// tracked by ASYNCcnt).  LDS byte offset = low 32 bits of the shared-aperture
// generic address (ISA 10.2).
__device__ __forceinline__ void async_ld16(const _Float16* g, _Float16* l) {
  unsigned lds_off = (unsigned)(uintptr_t)l;
  asm volatile("global_load_async_to_lds_b128 %0, %1, off"
               :: "v"(lds_off), "v"(g) : "memory");
}
__device__ __forceinline__ void wait_async0() {
  asm volatile("s_wait_asynccnt 0" ::: "memory");
}

// A-fragment (16x32 f16, M x K): lane l holds row (l&15); 16 halfs are
// K = [k0, k0+8) and [k0+16, k0+24) with k0 = (l>>4)*8.
__device__ __forceinline__ v16h frag_a(const _Float16* s, int rowBase,
                                       int lane, int stride) {
  int r  = rowBase + (lane & 15);
  int k0 = (lane >> 4) << 3;
  const _Float16* p = s + r * stride + k0;
  v8h lo = *(const v8h*)p;
  v8h hi = *(const v8h*)(p + 16);
  v16h f;
#pragma unroll
  for (int i = 0; i < 8; ++i) { f[i] = lo[i]; f[i + 8] = hi[i]; }
  return f;
}

// B-fragment (32x16 f16, K x N) from a [N][K] (transposed) LDS tile:
// lane l holds column (l&15); 16 halfs are K = [k0, k0+8) and [k0+16, k0+24) with k0 = (l>>4)*8.
__device__ __forceinline__ v16h frag_b(const _Float16* s, int colBase,
                                       int lane, int stride) {
  int n  = colBase + (lane & 15);
  int k0 = (lane >> 4) << 3;
  const _Float16* p = s + n * stride + k0;
  v8h lo = *(const v8h*)p;
  v8h hi = *(const v8h*)(p + 16);
  v16h f;
#pragma unroll
  for (int i = 0; i < 8; ++i) { f[i] = lo[i]; f[i + 8] = hi[i]; }
  return f;
}

// ---------------------------------------------------------------------------
// Weight transpose + f32 -> f16:  W[K][N] -> Wt[N][K]
// ---------------------------------------------------------------------------
__global__ void transpose_f16_kernel(const float* __restrict__ W,
                                     _Float16* __restrict__ Wt,
                                     int K, int N) {
  int id = blockIdx.x * 256 + threadIdx.x;
  if (id >= K * N) return;
  int k = id / N, n = id % N;
  Wt[(size_t)n * K + k] = (_Float16)W[id];
}

// ---------------------------------------------------------------------------
// LayerNorm over 384 dims, f32 in -> f16 out.  One wave per row.
// ---------------------------------------------------------------------------
__global__ __launch_bounds__(256) void ln_f16_kernel(
    const float* __restrict__ x, const float* __restrict__ g,
    const float* __restrict__ b, _Float16* __restrict__ out, int rows) {
  int wave = threadIdx.x >> 5, lane = threadIdx.x & 31;
  int row = blockIdx.x * 8 + wave;
  if (row >= rows) return;
  const float* xr = x + (size_t)row * CDIM;
  float v[12], s = 0.f, s2 = 0.f;
#pragma unroll
  for (int i = 0; i < 12; ++i) {
    float t = xr[lane + i * 32];
    v[i] = t; s += t; s2 += t * t;
  }
  s += SWZ_XOR(s, 1);  s2 += SWZ_XOR(s2, 1);
  s += SWZ_XOR(s, 2);  s2 += SWZ_XOR(s2, 2);
  s += SWZ_XOR(s, 4);  s2 += SWZ_XOR(s2, 4);
  s += SWZ_XOR(s, 8);  s2 += SWZ_XOR(s2, 8);
  s += SWZ_XOR(s, 16); s2 += SWZ_XOR(s2, 16);
  float mean = s * (1.f / CDIM);
  float var  = s2 * (1.f / CDIM) - mean * mean;
  float inv  = rsqrtf(var + 1e-5f);
  _Float16* orow = out + (size_t)row * CDIM;
#pragma unroll
  for (int i = 0; i < 12; ++i) {
    int c = lane + i * 32;
    orow[c] = (_Float16)((v[i] - mean) * inv * g[c] + b[c]);
  }
}

// ---------------------------------------------------------------------------
// DynamicPosBias MLP: 169 relative offsets -> [169,12] table (tiny, scalar).
// ---------------------------------------------------------------------------
__device__ __forceinline__ void ln_relu_24(const float* in, const float* g,
                                           const float* b, float* out) {
  float s = 0.f, s2 = 0.f;
#pragma unroll
  for (int i = 0; i < 24; ++i) { s += in[i]; s2 += in[i] * in[i]; }
  float m   = s * (1.f / 24.f);
  float var = s2 * (1.f / 24.f) - m * m;
  float inv = rsqrtf(var + 1e-5f);
#pragma unroll
  for (int i = 0; i < 24; ++i) {
    float t = (in[i] - m) * inv * g[i] + b[i];
    out[i] = t > 0.f ? t : 0.f;
  }
}

__global__ void pos_mlp_kernel(
    const float* __restrict__ ppw, const float* __restrict__ ppb,
    const float* __restrict__ l1g, const float* __restrict__ l1b,
    const float* __restrict__ p1w, const float* __restrict__ p1b,
    const float* __restrict__ l2g, const float* __restrict__ l2b,
    const float* __restrict__ p2w, const float* __restrict__ p2b,
    const float* __restrict__ l3g, const float* __restrict__ l3b,
    const float* __restrict__ p3w, const float* __restrict__ p3b,
    float* __restrict__ pos_tab) {
  int t = blockIdx.x * blockDim.x + threadIdx.x;
  if (t >= 169) return;
  float dh = (float)(t / 13) - 6.f;
  float dw = (float)(t % 13) - 6.f;
  float a[24], r[24];
#pragma unroll
  for (int j = 0; j < 24; ++j) a[j] = dh * ppw[j] + dw * ppw[24 + j] + ppb[j];
  ln_relu_24(a, l1g, l1b, r);
  for (int j = 0; j < 24; ++j) {
    float acc = p1b[j];
    for (int i = 0; i < 24; ++i) acc += r[i] * p1w[i * 24 + j];
    a[j] = acc;
  }
  ln_relu_24(a, l2g, l2b, r);
  for (int j = 0; j < 24; ++j) {
    float acc = p2b[j];
    for (int i = 0; i < 24; ++i) acc += r[i] * p2w[i * 24 + j];
    a[j] = acc;
  }
  ln_relu_24(a, l3g, l3b, r);
  for (int j = 0; j < 12; ++j) {
    float acc = p3b[j];
    for (int i = 0; i < 24; ++i) acc += r[i] * p3w[i * 12 + j];
    pos_tab[t * 12 + j] = acc;
  }
}

// Gather pos table -> bias[12][49][49]
__global__ void bias_gather_kernel(const float* __restrict__ pos_tab,
                                   float* __restrict__ biasf) {
  int id = blockIdx.x * 256 + threadIdx.x;
  if (id >= 12 * 49 * 49) return;
  int h = id / 2401, rem = id % 2401;
  int i = rem / 49, j = rem % 49;
  int idx = ((i / 7) - (j / 7) + 6) * 13 + ((i % 7) - (j % 7) + 6);
  biasf[id] = pos_tab[idx * 12 + h];
}

// ---------------------------------------------------------------------------
// WMMA GEMM: out[M,N] = A[M,K](f16) * Bt[N,K](f16)^T + bias, fused epilogue.
// 256 threads = 8 waves; block tile 128x128; 32x64 per wave; K-step 32.
// Double-buffered LDS filled by async global->LDS; 1 barrier / step.
// MODE 0: qkv scatter to [win][head][tok][d] (q scaled)
// MODE 1: proj  -> f32 out = shortcut + acc + bias
// MODE 2: fc1   -> f16 out = GELU(acc + bias)
// MODE 3: fc2   -> f32 out = x2 + acc + bias
// ---------------------------------------------------------------------------
template <int MODE>
__global__ __launch_bounds__(256) void gemm_wmma_kernel(
    const _Float16* __restrict__ A, const _Float16* __restrict__ Bt,
    const float* __restrict__ bias, int K,
    const float* __restrict__ addsrc, float* __restrict__ outf,
    _Float16* __restrict__ outh) {
  __shared__ _Float16 As[2][128 * 40];
  __shared__ _Float16 Bs[2][128 * 40];
  const int tid  = threadIdx.x;
  const int m0   = blockIdx.x * 128;
  const int n0   = blockIdx.y * 128;
  const int wave = tid >> 5, lane = tid & 31;
  const int wm = wave & 3;      // 4 row tiles of 32
  const int wn = wave >> 2;     // 2 col tiles of 64
  const int half = lane >> 4, ncol = lane & 15;

  // fixed per-thread 16B-chunk coordinates for tile staging
  const int r0  = tid >> 2;            // 0..63
  const int kc0 = (tid & 3) << 3;      // 0,8,16,24
  const _Float16* Ag0 = A  + (size_t)(m0 + r0)      * K + kc0;
  const _Float16* Ag1 = A  + (size_t)(m0 + r0 + 64) * K + kc0;
  const _Float16* Bg0 = Bt + (size_t)(n0 + r0)      * K + kc0;
  const _Float16* Bg1 = Bt + (size_t)(n0 + r0 + 64) * K + kc0;
  const int lo = r0 * 40 + kc0;
  const int hi = (r0 + 64) * 40 + kc0;

  v8f acc[2][4];
#pragma unroll
  for (int i = 0; i < 2; ++i)
#pragma unroll
    for (int j = 0; j < 4; ++j)
#pragma unroll
      for (int e = 0; e < 8; ++e) acc[i][j][e] = 0.f;

  // prologue: async-stage tile 0
  async_ld16(Ag0, &As[0][lo]);
  async_ld16(Ag1, &As[0][hi]);
  async_ld16(Bg0, &Bs[0][lo]);
  async_ld16(Bg1, &Bs[0][hi]);
  wait_async0();
  __syncthreads();

  const int nsteps = K >> 5;
  for (int st = 0; st < nsteps; ++st) {
    const _Float16* Ac = As[st & 1];
    const _Float16* Bc = Bs[st & 1];
    if (st + 1 < nsteps) {            // fire-and-forget fills of other buffer
      int kb = (st + 1) << 5;
      _Float16* An = (_Float16*)As[(st + 1) & 1];
      _Float16* Bn = (_Float16*)Bs[(st + 1) & 1];
      async_ld16(Ag0 + kb, &An[lo]);
      async_ld16(Ag1 + kb, &An[hi]);
      async_ld16(Bg0 + kb, &Bn[lo]);
      async_ld16(Bg1 + kb, &Bn[hi]);
    }
    v16h af[2], bf[4];
    af[0] = frag_a(Ac, wm * 32,      lane, 40);
    af[1] = frag_a(Ac, wm * 32 + 16, lane, 40);
#pragma unroll
    for (int j = 0; j < 4; ++j) bf[j] = frag_b(Bc, wn * 64 + j * 16, lane, 40);
#pragma unroll
    for (int i = 0; i < 2; ++i)
#pragma unroll
      for (int j = 0; j < 4; ++j)
        acc[i][j] = wmma32(af[i], bf[j], acc[i][j]);
    wait_async0();                    // next buffer landed (own wave)
    __syncthreads();                  // ... and for all waves
  }

#pragma unroll
  for (int i = 0; i < 2; ++i) {
#pragma unroll
    for (int e = 0; e < 8; ++e) {
      int m = m0 + wm * 32 + i * 16 + e + half * 8;
      int win = 0, tok = 0;
      if (MODE == 0) {
        int b = m / 784, pos = m % 784;
        int h_ = pos / 28, w_ = pos % 28;
        win = b * 16 + (h_ / 7) * 4 + (w_ / 7);
        tok = (h_ % 7) * 7 + (w_ % 7);
      }
#pragma unroll
      for (int j = 0; j < 4; ++j) {
        int n = n0 + wn * 64 + j * 16 + ncol;
        float v = acc[i][j][e] + bias[n];
        if (MODE == 0) {
          int s = n / CDIM, c = n % CDIM;
          int head = c >> 5, d = c & 31;
          if (s == 0) v *= SCALE_Q;
          outh[(size_t)s * QS + (((size_t)win * 12 + head) * 49 + tok) * 32 + d]
              = (_Float16)v;
        } else if (MODE == 1) {
          size_t idx = (size_t)m * CDIM + n;
          outf[idx] = addsrc[idx] + v;
        } else if (MODE == 2) {
          float gl = 0.5f * v * (1.0f + erff(v * 0.7071067811865476f));
          outh[(size_t)m * 1536 + n] = (_Float16)gl;
        } else {
          size_t idx = (size_t)m * CDIM + n;
          outf[idx] = addsrc[idx] + v;
        }
      }
    }
  }
}

// ---------------------------------------------------------------------------
// Window attention, one block (4 waves) per (window, head).
// S = q kT via WMMA (pad 49->64, K=32), bias/mask staged in LDS (branch-free
// softmax), O = P v via WMMA.  Output scattered to token order (f16).
// ---------------------------------------------------------------------------
__global__ __launch_bounds__(128) void attn_wmma_kernel(
    const _Float16* __restrict__ qkv, const float* __restrict__ biasf,
    _Float16* __restrict__ out) {
  __shared__ _Float16 qs[64 * 40];
  __shared__ _Float16 ks[64 * 40];
  __shared__ _Float16 vt[32 * 72];   // v transposed: [d][tok], tok padded to 64
  __shared__ _Float16 pl[64 * 72];   // softmax probs, f16
  __shared__ float    bs[64 * 64];   // padded bias + column mask

  const int tid = threadIdx.x;
  const int wid = blockIdx.x;
  const int win = wid / 12, head = wid % 12;
  const size_t base = ((size_t)win * 12 + head) * 49 * 32;
  const _Float16* q = qkv + base;
  const _Float16* k = qkv + (size_t)QS + base;
  const _Float16* v = qkv + (size_t)2 * QS + base;
  const float* bh = biasf + head * 2401;

  // stage q,k rows (pad 49..63 with zero)
  for (int c = tid; c < 256; c += 128) {
    int row = c >> 2, kc = (c & 3) << 3;
    v8h z;
#pragma unroll
    for (int i = 0; i < 8; ++i) z[i] = (_Float16)0.f;
    v8h qv = z, kv = z;
    if (row < 49) {
      qv = *(const v8h*)(q + row * 32 + kc);
      kv = *(const v8h*)(k + row * 32 + kc);
    }
    *(v8h*)&qs[row * 40 + kc] = qv;
    *(v8h*)&ks[row * 40 + kc] = kv;
  }
  // v transposed into [32][64]
  for (int c = tid; c < 2048; c += 128) {
    int d = c >> 6, t = c & 63;
    _Float16 val = (_Float16)0.f;
    if (t < 49) val = v[t * 32 + d];
    vt[d * 72 + t] = val;
  }
  // padded bias tile: [64][64]; cols >=49 masked, rows >=49 bias 0
  for (int c = tid; c < 4096; c += 128) {
    int m = c >> 6, n = c & 63;
    float bv = -30000.f;
    if (n < 49) bv = (m < 49) ? bh[m * 49 + n] : 0.f;
    bs[c] = bv;
  }
  __syncthreads();

  const int wave = tid >> 5, lane = tid & 31;
  const int half = lane >> 4, ncol = lane & 15;

  // S = q kT : each wave handles 16 rows x 64 cols, K = 32 (one WMMA each)
  v16h aq = frag_a(qs, wave * 16, lane, 40);
  v8f s[4];
#pragma unroll
  for (int nt = 0; nt < 4; ++nt) {
    v8f z;
#pragma unroll
    for (int i = 0; i < 8; ++i) z[i] = 0.f;
    v16h bk = frag_b(ks, nt * 16, lane, 40);
    s[nt] = wmma32(aq, bk, z);
  }

  // branch-free bias + softmax; each row lives in 16 lanes of a half-wave
#pragma unroll
  for (int e = 0; e < 8; ++e) {
    int m = wave * 16 + e + half * 8;
    const float* brow = &bs[m * 64 + ncol];
    float sv[4], mx = -3.0e38f;
#pragma unroll
    for (int nt = 0; nt < 4; ++nt) {
      float x = s[nt][e] + brow[nt * 16];
      sv[nt] = x;
      mx = fmaxf(mx, x);
    }
    mx = fmaxf(mx, SWZ_XOR(mx, 1));
    mx = fmaxf(mx, SWZ_XOR(mx, 2));
    mx = fmaxf(mx, SWZ_XOR(mx, 4));
    mx = fmaxf(mx, SWZ_XOR(mx, 8));
    float ss = 0.f;
#pragma unroll
    for (int nt = 0; nt < 4; ++nt) {
      float p = __expf(sv[nt] - mx);
      sv[nt] = p; ss += p;
    }
    ss += SWZ_XOR(ss, 1);
    ss += SWZ_XOR(ss, 2);
    ss += SWZ_XOR(ss, 4);
    ss += SWZ_XOR(ss, 8);
    float inv = 1.0f / ss;
#pragma unroll
    for (int nt = 0; nt < 4; ++nt) {
      pl[m * 72 + nt * 16 + ncol] = (_Float16)(sv[nt] * inv);
    }
  }
  __syncthreads();

  // O = P v : 16 rows x 32 cols per wave, K = 64 in two 32-steps
  v8f o0, o1;
#pragma unroll
  for (int i = 0; i < 8; ++i) { o0[i] = 0.f; o1[i] = 0.f; }
#pragma unroll
  for (int kk = 0; kk < 64; kk += 32) {
    v16h pa = frag_a(pl + kk, wave * 16, lane, 72);
    v16h b0 = frag_b(vt + kk, 0,  lane, 72);
    v16h b1 = frag_b(vt + kk, 16, lane, 72);
    o0 = wmma32(pa, b0, o0);
    o1 = wmma32(pa, b1, o1);
  }

  // scatter to token-order f16 [TOKENS, 384]
  int b = win >> 4, r = win & 15, wh = r >> 2, ww = r & 3;
#pragma unroll
  for (int e = 0; e < 8; ++e) {
    int m = wave * 16 + e + half * 8;
    if (m < 49) {
      int ih = m / 7, iw = m % 7;
      size_t tokidx = (size_t)b * 784 + (wh * 7 + ih) * 28 + (ww * 7 + iw);
      _Float16* dst = out + tokidx * CDIM + head * 32;
      dst[ncol]      = (_Float16)o0[e];
      dst[16 + ncol] = (_Float16)o1[e];
    }
  }
}

// ---------------------------------------------------------------------------
// Host-side launch sequence (graph-capture safe: only launches on `stream`).
// ---------------------------------------------------------------------------
extern "C" void kernel_launch(void* const* d_in, const int* in_sizes, int n_in,
                              void* d_out, int out_size, void* d_ws,
                              size_t ws_size, hipStream_t stream) {
  const float* x       = (const float*)d_in[0];
  const float* n1g     = (const float*)d_in[1];
  const float* n1b     = (const float*)d_in[2];
  const float* qkv_w   = (const float*)d_in[3];
  const float* qkv_b   = (const float*)d_in[4];
  const float* proj_w  = (const float*)d_in[5];
  const float* proj_b  = (const float*)d_in[6];
  const float* n2g     = (const float*)d_in[7];
  const float* n2b     = (const float*)d_in[8];
  const float* fc1_w   = (const float*)d_in[9];
  const float* fc1_b   = (const float*)d_in[10];
  const float* fc2_w   = (const float*)d_in[11];
  const float* fc2_b   = (const float*)d_in[12];
  const float* pp_w    = (const float*)d_in[13];
  const float* pp_b    = (const float*)d_in[14];
  const float* l1_g    = (const float*)d_in[15];
  const float* l1_b    = (const float*)d_in[16];
  const float* p1_w    = (const float*)d_in[17];
  const float* p1_b    = (const float*)d_in[18];
  const float* l2_g    = (const float*)d_in[19];
  const float* l2_b    = (const float*)d_in[20];
  const float* p2_w    = (const float*)d_in[21];
  const float* p2_b    = (const float*)d_in[22];
  const float* l3_g    = (const float*)d_in[23];
  const float* l3_b    = (const float*)d_in[24];
  const float* p3_w    = (const float*)d_in[25];
  const float* p3_b    = (const float*)d_in[26];

  char* ws = (char*)d_ws;
  size_t off = 0;
  auto alloc = [&](size_t bytes) {
    size_t o = off;
    off += (bytes + 255) & ~(size_t)255;
    return o;
  };
  const size_t o_wtqkv = alloc((size_t)1152 * 384 * 2);
  const size_t o_wtprj = alloc((size_t)384 * 384 * 2);
  const size_t o_wtfc1 = alloc((size_t)1536 * 384 * 2);
  const size_t o_wtfc2 = alloc((size_t)384 * 1536 * 2);
  const size_t o_ptab  = alloc((size_t)169 * 12 * 4);
  const size_t o_bias  = alloc((size_t)12 * 49 * 49 * 4);
  const size_t o_h16   = alloc((size_t)TOKENS * CDIM * 2);      // LN1 out; reused as LN2 out
  const size_t o_qkv   = alloc((size_t)3 * TOKENS * CDIM * 2);  // q|k|v sections
  const size_t o_attn  = alloc((size_t)TOKENS * CDIM * 2);      // contiguous after qkv
  const size_t o_x2    = alloc((size_t)TOKENS * CDIM * 4);
  // fc1 output (TOKENS x 1536 f16) exactly aliases qkv+attn regions (both dead)
  const size_t o_mid   = o_qkv;

  _Float16* wt_qkv = (_Float16*)(ws + o_wtqkv);
  _Float16* wt_prj = (_Float16*)(ws + o_wtprj);
  _Float16* wt_fc1 = (_Float16*)(ws + o_wtfc1);
  _Float16* wt_fc2 = (_Float16*)(ws + o_wtfc2);
  float*    ptab   = (float*)(ws + o_ptab);
  float*    biasf  = (float*)(ws + o_bias);
  _Float16* h16    = (_Float16*)(ws + o_h16);
  _Float16* qkv16  = (_Float16*)(ws + o_qkv);
  _Float16* attn16 = (_Float16*)(ws + o_attn);
  float*    x2     = (float*)(ws + o_x2);
  _Float16* mid16  = (_Float16*)(ws + o_mid);
  (void)o_attn; (void)ws_size; (void)n_in; (void)in_sizes; (void)out_size;

  // 1) weight transposes (f32 -> f16, [K,N] -> [N,K])
  transpose_f16_kernel<<<(384 * 1152 + 255) / 256, 256, 0, stream>>>(qkv_w, wt_qkv, 384, 1152);
  transpose_f16_kernel<<<(384 * 384 + 255) / 256, 256, 0, stream>>>(proj_w, wt_prj, 384, 384);
  transpose_f16_kernel<<<(384 * 1536 + 255) / 256, 256, 0, stream>>>(fc1_w, wt_fc1, 384, 1536);
  transpose_f16_kernel<<<(1536 * 384 + 255) / 256, 256, 0, stream>>>(fc2_w, wt_fc2, 1536, 384);

  // 2) dynamic position bias
  pos_mlp_kernel<<<1, 192, 0, stream>>>(pp_w, pp_b, l1_g, l1_b, p1_w, p1_b,
                                        l2_g, l2_b, p2_w, p2_b, l3_g, l3_b,
                                        p3_w, p3_b, ptab);
  bias_gather_kernel<<<(12 * 49 * 49 + 255) / 256, 256, 0, stream>>>(ptab, biasf);

  // 3) LN1 -> f16
  ln_f16_kernel<<<TOKENS / 8, 256, 0, stream>>>(x, n1g, n1b, h16, TOKENS);

  // 4) QKV GEMM (scatter into window/head layout)
  gemm_wmma_kernel<0><<<dim3(TOKENS / 128, 1152 / 128), 256, 0, stream>>>(
      h16, wt_qkv, qkv_b, 384, nullptr, nullptr, qkv16);

  // 5) window attention
  attn_wmma_kernel<<<NPAIRS, 128, 0, stream>>>(qkv16, biasf, attn16);

  // 6) proj + residual -> x2 (f32)
  gemm_wmma_kernel<1><<<dim3(TOKENS / 128, 384 / 128), 256, 0, stream>>>(
      attn16, wt_prj, proj_b, 384, x, x2, nullptr);

  // 7) LN2 -> f16 (reuse h16)
  ln_f16_kernel<<<TOKENS / 8, 256, 0, stream>>>(x2, n2g, n2b, h16, TOKENS);

  // 8) fc1 + exact GELU -> mid16
  gemm_wmma_kernel<2><<<dim3(TOKENS / 128, 1536 / 128), 256, 0, stream>>>(
      h16, wt_fc1, fc1_b, 384, nullptr, nullptr, mid16);

  // 9) fc2 + residual -> d_out (f32)
  gemm_wmma_kernel<3><<<dim3(TOKENS / 128, 384 / 128), 256, 0, stream>>>(
      mid16, wt_fc2, fc2_b, 1536, x2, (float*)d_out, nullptr);
}
